// SeqAttentionBlock_81114752353008
// MI455X (gfx1250) — compile-verified, matrix operand order checked by reading, not run on hardware
//
#include <hip/hip_runtime.h>
#include <hip/hip_bf16.h>
#include <math.h>

// SeqAttentionBlock for MI455X (gfx1250, wave32, WMMA).
// B=4 T=512 M=16 D=128 P=128 H=4 E=32.
// Stage 1: bf16 WMMA projections -> workspace QC[b,m,h,t,64] (q|q_t),
//          KC[b,m,h,t,64] (k|k_t), Vt[b,m,h,e,t]  (V stored TRANSPOSED so
//          the attention-stage B fragments are contiguous b128 loads).
// Stage 2: flash attention per (b,m,h): scores via K=64 bf16 WMMA chain
//          (q.kT + qt.ktT fused by feature concat), online softmax with
//          half-wave shfl reductions, P*V via WMMA with P relaid via LDS.
// Workspace: 16MB + 16MB + 8MB = 40MB bf16.

#define B_ 4
#define T_ 512
#define M_ 16
#define D_ 128
#define P_ 128
#define H_ 4
#define E_ 32

typedef __attribute__((ext_vector_type(16))) __bf16 v16bf;
typedef __attribute__((ext_vector_type(8)))  float  v8f;

static __device__ __forceinline__ v8f wmma_bf16(v16bf a, v16bf b, v8f c) {
  // D = A(16x32) * B(32x16) + C, f32 accumulate
  return __builtin_amdgcn_wmma_f32_16x16x32_bf16(
      /*neg_a=*/false, a, /*neg_b=*/false, b,
      /*c_mod=*/(short)0, c, /*reuse_a=*/false, /*reuse_b=*/false);
}

static __device__ __forceinline__ float half16_max(float v) {
  // reduce across a 16-lane half (rows split lanes 0-15 / 16-31 in C layout)
  #pragma unroll
  for (int m = 1; m < 16; m <<= 1) v = fmaxf(v, __shfl_xor(v, m, 32));
  return v;
}
static __device__ __forceinline__ float half16_sum(float v) {
  #pragma unroll
  for (int m = 1; m < 16; m <<= 1) v += __shfl_xor(v, m, 32);
  return v;
}

// ---------------------------------------------------------------------------
// Kernel 1: projections.  One wave = one 16x16 output tile of one GEMM.
// Tile id encodes: widx(5: q,k,v,qt,kt) x m(16) x rowtile(128) x coltile(8).
// ---------------------------------------------------------------------------
__global__ __launch_bounds__(256)
void proj_kernel(const float* __restrict__ inp, const float* __restrict__ pos,
                 const float* __restrict__ Wq,  const float* __restrict__ Bq,
                 const float* __restrict__ Wk,  const float* __restrict__ Bk,
                 const float* __restrict__ Wv,  const float* __restrict__ Bv,
                 const float* __restrict__ Wqt, const float* __restrict__ Bqt,
                 const float* __restrict__ Wkt, const float* __restrict__ Bkt,
                 __bf16* __restrict__ QC, __bf16* __restrict__ KC,
                 __bf16* __restrict__ Vt)
{
  const int wave = threadIdx.x >> 5;
  const int l    = threadIdx.x & 31;
  const int sel  = l & 15;          // A: row index; B: col index
  const int hi   = l >> 4;
  const int kb   = hi * 8;          // K sub-block base per ISA 16-bit layout

  const int tile = blockIdx.x * 8 + wave;     // 81920 tiles total
  const int ct   = tile & 7;                  // col tile (P/16)
  const int rt   = (tile >> 3) & 127;         // row tile (B*T/16)
  const int m    = (tile >> 10) & 15;
  const int widx = tile >> 14;                // 0..4

  const float* W; const float* Bi;
  switch (widx) {
    case 0: W = Wq;  Bi = Bq;  break;
    case 1: W = Wk;  Bi = Bk;  break;
    case 2: W = Wv;  Bi = Bv;  break;
    case 3: W = Wqt; Bi = Bqt; break;
    default: W = Wkt; Bi = Bkt; break;
  }
  const int p0 = ct * 16;

  // A source row for this lane (shared by all K chunks)
  const int rgA = rt * 16 + sel;                       // flat (b*T+t)
  const float* xrow = (widx < 3) ? (inp + ((size_t)rgA * M_ + m) * D_)
                                 : (pos + (size_t)rgA * D_);
  // B source row (column p0+sel of W^T == row p0+sel of W[m])
  const float* wrow = W + ((size_t)m * P_ + p0 + sel) * D_;

  v8f c = {};
  #pragma unroll
  for (int d0 = 0; d0 < D_; d0 += 32) {
    v16bf a, bfr;
    #pragma unroll
    for (int j = 0; j < 8; ++j) {
      a[j]       = (__bf16)xrow[d0 + kb + j];
      a[8 + j]   = (__bf16)xrow[d0 + 16 + kb + j];
      bfr[j]     = (__bf16)wrow[d0 + kb + j];
      bfr[8 + j] = (__bf16)wrow[d0 + 16 + kb + j];
    }
    c = wmma_bf16(a, bfr, c);
  }

  const int   p    = p0 + sel;
  const int   h    = p >> 5;
  const int   e    = p & 31;
  const float bias = Bi[m * P_ + p];

  #pragma unroll
  for (int r = 0; r < 8; ++r) {
    const int i   = r + 8 * hi;               // row within tile (C/D layout)
    const int rg  = rt * 16 + i;
    const int b   = rg >> 9;
    const int t   = rg & 511;
    const float val = c[r] + bias;
    const size_t bmh = ((size_t)(b * M_ + m) * H_ + h);
    if (widx == 0)      QC[(bmh * T_ + t) * 64 + e]      = (__bf16)val;
    else if (widx == 3) QC[(bmh * T_ + t) * 64 + 32 + e] = (__bf16)val;
    else if (widx == 1) KC[(bmh * T_ + t) * 64 + e]      = (__bf16)val;
    else if (widx == 4) KC[(bmh * T_ + t) * 64 + 32 + e] = (__bf16)val;
    else                Vt[(bmh * E_ + e) * T_ + t]      = (__bf16)val;
  }
}

// ---------------------------------------------------------------------------
// Kernel 2: flash attention per (b,m,h). 4 waves x 16 query rows per block.
// ---------------------------------------------------------------------------
__global__ __launch_bounds__(128)
void attn_kernel(const __bf16* __restrict__ QC, const __bf16* __restrict__ KC,
                 const __bf16* __restrict__ Vt,
                 const unsigned char* __restrict__ mask,
                 float* __restrict__ out)
{
  __shared__ __bf16 ldsP[4][16 * 32];   // per-wave P tile (16 rows x 32 s)

  const int wave = threadIdx.x >> 5;
  const int l    = threadIdx.x & 31;
  const int sel  = l & 15;
  const int hi   = l >> 4;
  const int kb   = hi * 8;

  const int bmh = blockIdx.x;           // (b*16+m)*4+h, 0..255
  const int b   = bmh >> 6;
  const int m   = (bmh >> 2) & 15;
  const int t0  = blockIdx.y * 64 + wave * 16;

  const __bf16* qc = QC + (size_t)bmh * T_ * 64;
  const __bf16* kc = KC + (size_t)bmh * T_ * 64;
  const __bf16* vt = Vt + (size_t)bmh * E_ * T_;
  const unsigned char* km = mask + (size_t)b * T_ * M_ + m;

  // V^T rows for this lane's output columns (fixed e, contiguous in s)
  const __bf16* vrow0 = vt + (size_t)sel * T_;         // e = sel
  const __bf16* vrow1 = vt + (size_t)(16 + sel) * T_;  // e = 16+sel

  // Q fragment (16 rows x 64 feat = two k=32 A fragments), fixed per wave
  v16bf aq0, aq1;
  {
    const __bf16* qrow = qc + (size_t)(t0 + sel) * 64;
    #pragma unroll
    for (int j = 0; j < 8; ++j) {
      aq0[j]     = qrow[kb + j];       aq0[8 + j] = qrow[16 + kb + j];
      aq1[j]     = qrow[32 + kb + j];  aq1[8 + j] = qrow[48 + kb + j];
    }
  }

  float mrow[8], lrow[8];
  v8f o0 = {}, o1 = {};
  #pragma unroll
  for (int r = 0; r < 8; ++r) { mrow[r] = -__builtin_inff(); lrow[r] = 0.0f; }

  const float scale = 0.08838834764831845f;   // 1/(2*sqrt(32))
  const int smax = t0 + 15;                   // causal limit for this wave

  for (int s0 = 0; s0 <= smax; s0 += 32) {
    if (s0 + 32 <= smax) {                    // prefetch next K/V chunk
      __builtin_prefetch(kc + (size_t)(s0 + 32 + sel) * 64, 0, 0);
      __builtin_prefetch(vrow0 + s0 + 32, 0, 0);
    }
    // K^T fragments: B(k,n) = KC[s0+n][k]; lane's column n = sel
    const __bf16* krow0 = kc + (size_t)(s0 + sel) * 64;
    const __bf16* krow1 = kc + (size_t)(s0 + 16 + sel) * 64;
    v16bf b00, b01, b10, b11;
    #pragma unroll
    for (int j = 0; j < 8; ++j) {
      b00[j] = krow0[kb + j];       b00[8 + j] = krow0[16 + kb + j];
      b01[j] = krow0[32 + kb + j];  b01[8 + j] = krow0[48 + kb + j];
      b10[j] = krow1[kb + j];       b10[8 + j] = krow1[16 + kb + j];
      b11[j] = krow1[32 + kb + j];  b11[8 + j] = krow1[48 + kb + j];
    }
    v8f c0 = {}, c1 = {};
    c0 = wmma_bf16(aq0, b00, c0);  c0 = wmma_bf16(aq1, b01, c0);
    c1 = wmma_bf16(aq0, b10, c1);  c1 = wmma_bf16(aq1, b11, c1);

    const int s_lo = s0 + sel, s_hi = s0 + 16 + sel;
    const bool km_lo = km[(size_t)s_lo * M_] != 0;
    const bool km_hi = km[(size_t)s_hi * M_] != 0;

    #pragma unroll
    for (int r = 0; r < 8; ++r) {
      const int i = r + 8 * hi;               // row within tile
      const int t = t0 + i;
      float x0 = (km_lo && s_lo <= t) ? c0[r] * scale : -__builtin_inff();
      float x1 = (km_hi && s_hi <= t) ? c1[r] * scale : -__builtin_inff();
      float mx = half16_max(fmaxf(x0, x1));
      float nm = fmaxf(mrow[r], mx);
      float al = __expf(mrow[r] - nm);
      float p0 = __expf(x0 - nm);
      float p1 = __expf(x1 - nm);
      lrow[r] = lrow[r] * al + half16_sum(p0 + p1);
      mrow[r] = nm;
      o0[r] *= al;  o1[r] *= al;
      ldsP[wave][i * 32 + sel]      = (__bf16)p0;
      ldsP[wave][i * 32 + 16 + sel] = (__bf16)p1;
    }

    // P fragment (16x32 A) from LDS; V fragments from transposed V rows
    // (contiguous in s -> vector b128 loads)
    v16bf ap, bv0, bv1;
    const __bf16* prow = &ldsP[wave][sel * 32];
    #pragma unroll
    for (int j = 0; j < 8; ++j) {
      ap[j]      = prow[kb + j];
      ap[8 + j]  = prow[16 + kb + j];
      bv0[j]     = vrow0[s0 + kb + j];
      bv0[8 + j] = vrow0[s0 + 16 + kb + j];
      bv1[j]     = vrow1[s0 + kb + j];
      bv1[8 + j] = vrow1[s0 + 16 + kb + j];
    }
    o0 = wmma_bf16(ap, bv0, o0);
    o1 = wmma_bf16(ap, bv1, o1);
  }

  const int h = bmh & 3;
  #pragma unroll
  for (int r = 0; r < 8; ++r) {
    const int i = r + 8 * hi;
    const int t = t0 + i;
    const float inv = (lrow[r] > 0.0f) ? (1.0f / lrow[r]) : 0.0f;
    float* orow = out + (((size_t)(b * T_ + t) * M_ + m) * P_) + h * E_;
    orow[sel]      = o0[r] * inv;
    orow[16 + sel] = o1[r] * inv;
  }
}

// ---------------------------------------------------------------------------
extern "C" void kernel_launch(void* const* d_in, const int* in_sizes, int n_in,
                              void* d_out, int out_size, void* d_ws, size_t ws_size,
                              hipStream_t stream) {
  const float* inp = (const float*)d_in[0];
  const float* pos = (const float*)d_in[1];
  const unsigned char* mask = (const unsigned char*)d_in[2];
  const float* Wq  = (const float*)d_in[3];
  const float* Bq  = (const float*)d_in[4];
  const float* Wk  = (const float*)d_in[5];
  const float* Bk  = (const float*)d_in[6];
  const float* Wv  = (const float*)d_in[7];
  const float* Bv  = (const float*)d_in[8];
  const float* Wqt = (const float*)d_in[9];
  const float* Bqt = (const float*)d_in[10];
  const float* Wkt = (const float*)d_in[11];
  const float* Bkt = (const float*)d_in[12];

  // Workspace layout (bf16): QC 8Mi elems, KC 8Mi elems, Vt 4Mi elems = 40MB
  __bf16* QC = (__bf16*)d_ws;
  __bf16* KC = QC + (size_t)B_ * M_ * H_ * T_ * 64;
  __bf16* Vt = KC + (size_t)B_ * M_ * H_ * T_ * 64;

  // 81920 wave-tiles / 8 waves per 256-thread block
  proj_kernel<<<10240, 256, 0, stream>>>(inp, pos, Wq, Bq, Wk, Bk, Wv, Bv,
                                         Wqt, Bqt, Wkt, Bkt, QC, KC, Vt);
  attn_kernel<<<dim3(256, 8), 128, 0, stream>>>(QC, KC, Vt, mask, (float*)d_out);
}
